// AudioMambaModel_19559281066130
// MI455X (gfx1250) — compile-verified, matrix operand order checked by reading, not run on hardware
//
#include <hip/hip_runtime.h>
#include <hip/hip_bf16.h>
#include <math.h>

typedef __attribute__((ext_vector_type(16))) _Float16 v16h;
typedef __attribute__((ext_vector_type(8)))  float    v8f;

#define BATCH    4
#define L_SEQ    256      // (128/16)*(512/16)
#define D_MODEL  512
#define D_INNER  1024
#define D_STATE  16
#define DT_RANK  32
#define DEPTH    4
#define NTOK     (BATCH * L_SEQ)   // 1024 token rows

__device__ __forceinline__ float apply_act(float v, int ACT) {
  if (ACT == 2) v = (v > 20.f) ? v : logf(1.f + __expf(v));  // softplus
  if (ACT == 3) v = fmaxf(v, 0.f);                           // relu
  return v;
}

// ---------------------------------------------------------------------------
// Fast-path WMMA GEMM (requires M%32==0, N%16==0, K%32==0).
// Each wave computes a 32x16 output tile: two 16x16 WMMAs sharing one B tile.
// No lane-divergent guards -> EXEC stays all-ones through the whole K loop.
// C[M,N] = act( A[M,K] * W[N,K]^T + bias[N] ) (+ C if ACCUM)
// ---------------------------------------------------------------------------
template <int ACT, bool ACCUM>
__global__ void wmma_gemm_fast(const float* __restrict__ A, int lda,
                               const float* __restrict__ W, int ldw,
                               const float* __restrict__ bias,
                               float* __restrict__ C, int ldc,
                               int N, int K) {
  const int lane = threadIdx.x & 31;
  const int wave = threadIdx.x >> 5;
  const int tileN = blockIdx.x * 4 + wave;
  const int n0 = tileN * 16;
  if (n0 >= N) return;                     // wave-uniform: EXEC stays full
  const int m0 = blockIdx.y * 32;
  const int hh = lane >> 4;                // K-striping half per ISA layout
  const int li = lane & 15;

  const float* __restrict__ arow0 = A + (size_t)(m0 + li) * lda;
  const float* __restrict__ arow1 = arow0 + (size_t)16 * lda;
  const float* __restrict__ wrow  = W + (size_t)(n0 + li) * ldw;

  v8f acc0 = {};
  v8f acc1 = {};
  for (int k0 = 0; k0 < K; k0 += 32) {
    const int kb = k0 + hh * 8;            // runs: [kb, kb+7] and [kb+16, kb+23]
    v16h a0, a1, bv;
#pragma unroll
    for (int e = 0; e < 8; ++e) {
      a0[e]     = (_Float16)arow0[kb + e];
      a0[8 + e] = (_Float16)arow0[kb + 16 + e];
      a1[e]     = (_Float16)arow1[kb + e];
      a1[8 + e] = (_Float16)arow1[kb + 16 + e];
      bv[e]     = (_Float16)wrow[kb + e];
      bv[8 + e] = (_Float16)wrow[kb + 16 + e];
    }
    acc0 = __builtin_amdgcn_wmma_f32_16x16x32_f16(false, a0, false, bv,
                                                  (short)0, acc0, false, false);
    acc1 = __builtin_amdgcn_wmma_f32_16x16x32_f16(false, a1, false, bv,
                                                  (short)0, acc1, false, false);
  }

  const int ncol = n0 + li;
  float bval = 0.f;
  if (ACT >= 1) bval = bias[ncol];
#pragma unroll
  for (int r = 0; r < 8; ++r) {
    const int mm = m0 + hh * 8 + r;        // C layout: VGPR r -> M = r + 8*half
    float v0 = apply_act(acc0[r] + bval, ACT);
    float v1 = apply_act(acc1[r] + bval, ACT);
    float* cp0 = C + (size_t)mm * ldc + ncol;
    float* cp1 = C + (size_t)(mm + 16) * ldc + ncol;
    if (ACCUM) { v0 += *cp0; v1 += *cp1; }
    *cp0 = v0;
    *cp1 = v1;
  }
}

// ---------------------------------------------------------------------------
// Ragged WMMA GEMM (handles arbitrary M, N with zero-fill loads / guarded
// stores). Used only for the tiny classifier head (M=4, N=512/10).
// ---------------------------------------------------------------------------
template <int ACT, bool ACCUM>
__global__ void wmma_gemm_ragged(const float* __restrict__ A, int lda,
                                 const float* __restrict__ W, int ldw,
                                 const float* __restrict__ bias,
                                 float* __restrict__ C, int ldc,
                                 int M, int N, int K) {
  const int lane = threadIdx.x & 31;
  const int wave = threadIdx.x >> 5;
  const int tileN = blockIdx.x * 4 + wave;
  const int tileM = blockIdx.y;
  const int hh = lane >> 4;
  const int li = lane & 15;
  const int m0 = tileM * 16, n0 = tileN * 16;
  const int mrow = m0 + li;
  const int ncol = n0 + li;
  const bool mOK = (mrow < M);
  const bool nOK = (ncol < N);
  const float* __restrict__ arow = A + (size_t)mrow * lda;
  const float* __restrict__ wrow = W + (size_t)ncol * ldw;

  v8f acc = {};
  for (int k0 = 0; k0 < K; k0 += 32) {
    v16h av = {};
    v16h bv = {};
    const int kb = k0 + hh * 8;
    if (mOK) {
#pragma unroll
      for (int e = 0; e < 8; ++e) {
        av[e]     = (_Float16)arow[kb + e];
        av[8 + e] = (_Float16)arow[kb + 16 + e];
      }
    }
    if (nOK) {
#pragma unroll
      for (int e = 0; e < 8; ++e) {
        bv[e]     = (_Float16)wrow[kb + e];
        bv[8 + e] = (_Float16)wrow[kb + 16 + e];
      }
    }
    acc = __builtin_amdgcn_wmma_f32_16x16x32_f16(false, av, false, bv,
                                                 (short)0, acc, false, false);
  }

  if (!nOK) return;
  float bval = 0.f;
  if (ACT >= 1) bval = bias[ncol];
#pragma unroll
  for (int r = 0; r < 8; ++r) {
    const int mm = m0 + hh * 8 + r;
    if (mm >= M) continue;
    float v = apply_act(acc[r] + bval, ACT);
    float* cp = C + (size_t)mm * ldc + ncol;
    if (ACCUM) v += *cp;
    *cp = v;
  }
}

template <int ACT, bool ACCUM>
static inline void launch_gemm(const float* A, int lda, const float* W, int ldw,
                               const float* bias, float* C, int ldc,
                               int M, int N, int K, hipStream_t s) {
  if ((M % 32 == 0) && (N % 16 == 0)) {
    dim3 block(128);
    dim3 grid((unsigned)((N / 16 + 3) / 4), (unsigned)(M / 32));
    wmma_gemm_fast<ACT, ACCUM><<<grid, block, 0, s>>>(A, lda, W, ldw, bias, C, ldc, N, K);
  } else {
    dim3 block(128);
    dim3 grid((unsigned)(((N + 15) / 16 + 3) / 4), (unsigned)((M + 15) / 16));
    wmma_gemm_ragged<ACT, ACCUM><<<grid, block, 0, s>>>(A, lda, W, ldw, bias, C, ldc, M, N, K);
  }
}

// ---------------------------------------------------------------------------
// im2row for the 16x16 patch embed: Ap[tok][p*16+q] = x[b][hp*16+p][wp*16+q]
// ---------------------------------------------------------------------------
__global__ void im2row_kernel(const float* __restrict__ x, float* __restrict__ Ap) {
  int idx = blockIdx.x * blockDim.x + threadIdx.x;   // NTOK*256
  if (idx >= NTOK * 256) return;
  int kk = idx & 255, t = idx >> 8;
  int b = t >> 8, hw = t & 255;
  int hp = hw >> 5, wp = hw & 31;
  int p = kk >> 4, q = kk & 15;
  Ap[idx] = x[((size_t)(b * 128 + hp * 16 + p)) * 512 + wp * 16 + q];
}

// ---------------------------------------------------------------------------
// LayerNorm over D_MODEL=512; one wave per row, 16 elems/lane, shfl reduction
// ---------------------------------------------------------------------------
__global__ void layernorm_kernel(const float* __restrict__ in,
                                 const float* __restrict__ w,
                                 const float* __restrict__ b,
                                 float* __restrict__ out, int rows) {
  int wave = threadIdx.x >> 5, lane = threadIdx.x & 31;
  int row = blockIdx.x * (blockDim.x >> 5) + wave;
  if (row >= rows) return;
  const float* r = in + (size_t)row * D_MODEL;
  float v[16];
  float s = 0.f;
#pragma unroll
  for (int j = 0; j < 16; ++j) { v[j] = r[lane + j * 32]; s += v[j]; }
#pragma unroll
  for (int off = 16; off; off >>= 1) s += __shfl_xor(s, off);
  const float mean = s * (1.f / D_MODEL);
  float vs = 0.f;
#pragma unroll
  for (int j = 0; j < 16; ++j) { float d = v[j] - mean; vs += d * d; }
#pragma unroll
  for (int off = 16; off; off >>= 1) vs += __shfl_xor(vs, off);
  const float inv = rsqrtf(vs * (1.f / D_MODEL) + 1e-5f);
#pragma unroll
  for (int j = 0; j < 16; ++j) {
    int d = lane + j * 32;
    out[(size_t)row * D_MODEL + d] = (v[j] - mean) * inv * w[d] + b[d];
  }
}

// ---------------------------------------------------------------------------
// Depthwise causal conv (K=4) + SiLU. dir=0: u[l]=sum_k w[k]*xc[l-3+k]
// dir=1 (flipped stream, output in original order): u[l]=sum_k w[k]*xc[l+3-k]
// xc = first D_INNER columns of xz (row stride 2*D_INNER)
// ---------------------------------------------------------------------------
__global__ void conv_silu_kernel(const float* __restrict__ xz,
                                 const float* __restrict__ cw,
                                 const float* __restrict__ cb,
                                 float* __restrict__ u, int dir) {
  int idx = blockIdx.x * blockDim.x + threadIdx.x;   // NTOK*D_INNER
  if (idx >= NTOK * D_INNER) return;
  int d = idx & (D_INNER - 1);
  int tok = idx >> 10;
  int b = tok >> 8, l = tok & 255;
  const float* w = cw + (size_t)d * 4;
  float acc = cb[d];
#pragma unroll
  for (int k = 0; k < 4; ++k) {
    int ls = dir ? (l + 3 - k) : (l - 3 + k);
    if (ls >= 0 && ls < L_SEQ)
      acc += w[k] * xz[(size_t)(b * L_SEQ + ls) * (2 * D_INNER) + d];
  }
  u[idx] = acc / (1.f + __expf(-acc));   // silu
}

// ---------------------------------------------------------------------------
// Selective scan. 16 lanes (s=0..15) per (b,d) pair, 2 pairs per wave.
// h[l] = h[prev]*exp(dt*A[d,s]) + dt*u*B[l,s];  y[l,d] = sum_s h*C[l,s] + u*D
// dir=1 runs the recurrence from l=L-1 down to 0 (backward direction).
// ---------------------------------------------------------------------------
template <bool ACCUM>
__global__ void scan_kernel(const float* __restrict__ dt,
                            const float* __restrict__ u,
                            const float* __restrict__ xdbl,
                            const float* __restrict__ A_log,
                            const float* __restrict__ Dp,
                            float* __restrict__ y, int dir) {
  int s = threadIdx.x & 15;
  int pair = blockIdx.x * (blockDim.x >> 4) + (threadIdx.x >> 4);
  if (pair >= BATCH * D_INNER) return;
  int b = pair >> 10;
  int d = pair & (D_INNER - 1);
  const float Aval = -__expf(A_log[(size_t)d * D_STATE + s]);
  const float Dv = Dp[d];
  float h = 0.f;
  for (int i = 0; i < L_SEQ; ++i) {
    int l = dir ? (L_SEQ - 1 - i) : i;
    size_t tok = (size_t)(b * L_SEQ + l);
    float dtv = dt[tok * D_INNER + d];
    float uv  = u[tok * D_INNER + d];
    float Bv  = xdbl[tok * 64 + DT_RANK + s];
    float Cv  = xdbl[tok * 64 + DT_RANK + D_STATE + s];
    h = h * __expf(dtv * Aval) + dtv * uv * Bv;
    float contrib = h * Cv;
#pragma unroll
    for (int off = 8; off; off >>= 1) contrib += __shfl_xor(contrib, off);
    if (s == 0) {
      float out = contrib + uv * Dv;
      if (ACCUM) out += y[tok * D_INNER + d];
      y[tok * D_INNER + d] = out;
    }
  }
}

// ---------------------------------------------------------------------------
// Gate: y *= silu(z), z = columns [D_INNER, 2*D_INNER) of xz
// ---------------------------------------------------------------------------
__global__ void gate_kernel(float* __restrict__ y, const float* __restrict__ xz) {
  int idx = blockIdx.x * blockDim.x + threadIdx.x;
  if (idx >= NTOK * D_INNER) return;
  int d = idx & (D_INNER - 1);
  int tok = idx >> 10;
  float z = xz[(size_t)tok * (2 * D_INNER) + D_INNER + d];
  y[idx] *= z / (1.f + __expf(-z));
}

// ---------------------------------------------------------------------------
// Mean over L then LayerNorm over D_MODEL (classifier pre-norm). 1 block / batch.
// ---------------------------------------------------------------------------
__global__ void pool_ln_kernel(const float* __restrict__ hn,
                               const float* __restrict__ lw,
                               const float* __restrict__ lb,
                               float* __restrict__ cvec) {
  __shared__ float red[256];
  int b = blockIdx.x, t = threadIdx.x;
  float f0 = 0.f, f1 = 0.f;
  for (int l = 0; l < L_SEQ; ++l) {
    size_t base = (size_t)(b * L_SEQ + l) * D_MODEL;
    f0 += hn[base + t];
    f1 += hn[base + t + 256];
  }
  f0 *= (1.f / L_SEQ);
  f1 *= (1.f / L_SEQ);
  red[t] = f0 + f1;
  __syncthreads();
  for (int o = 128; o; o >>= 1) { if (t < o) red[t] += red[t + o]; __syncthreads(); }
  const float mean = red[0] * (1.f / D_MODEL);
  __syncthreads();
  float d0 = f0 - mean, d1 = f1 - mean;
  red[t] = d0 * d0 + d1 * d1;
  __syncthreads();
  for (int o = 128; o; o >>= 1) { if (t < o) red[t] += red[t + o]; __syncthreads(); }
  const float inv = rsqrtf(red[0] * (1.f / D_MODEL) + 1e-5f);
  cvec[(size_t)b * D_MODEL + t]       = d0 * inv * lw[t] + lb[t];
  cvec[(size_t)b * D_MODEL + t + 256] = d1 * inv * lw[t + 256] + lb[t + 256];
}

// ---------------------------------------------------------------------------
extern "C" void kernel_launch(void* const* d_in, const int* in_sizes, int n_in,
                              void* d_out, int out_size, void* d_ws, size_t ws_size,
                              hipStream_t stream) {
  (void)in_sizes; (void)n_in; (void)out_size; (void)ws_size;
  const float* x         = (const float*)d_in[0];
  const float* patch_w   = (const float*)d_in[1];
  const float* patch_b   = (const float*)d_in[2];
  const float* in_proj_w = (const float*)d_in[3];
  const float* conv_w_f  = (const float*)d_in[4];
  const float* conv_b_f  = (const float*)d_in[5];
  const float* xproj_w_f = (const float*)d_in[6];
  const float* dtproj_w_f= (const float*)d_in[7];
  const float* dtproj_b_f= (const float*)d_in[8];
  const float* A_log_f   = (const float*)d_in[9];
  const float* D_f       = (const float*)d_in[10];
  const float* conv_w_b  = (const float*)d_in[11];
  const float* conv_b_b  = (const float*)d_in[12];
  const float* xproj_w_b = (const float*)d_in[13];
  const float* dtproj_w_b= (const float*)d_in[14];
  const float* dtproj_b_b= (const float*)d_in[15];
  const float* A_log_b   = (const float*)d_in[16];
  const float* D_b       = (const float*)d_in[17];
  const float* out_proj_w= (const float*)d_in[18];
  const float* norm_w    = (const float*)d_in[19];
  const float* norm_b    = (const float*)d_in[20];
  const float* normf_w   = (const float*)d_in[21];
  const float* normf_b   = (const float*)d_in[22];
  const float* ln_w      = (const float*)d_in[23];
  const float* ln_b      = (const float*)d_in[24];
  const float* fc1_w     = (const float*)d_in[25];
  const float* fc1_b     = (const float*)d_in[26];
  const float* fc2_w     = (const float*)d_in[27];
  const float* fc2_b     = (const float*)d_in[28];

  // workspace carve-up (floats)
  float* ws   = (float*)d_ws;
  float* Ap   = ws;                 // NTOK x 256         (262144)
  float* h    = Ap  + 262144;       // NTOK x 512         (524288)
  float* xn   = h   + 524288;       // NTOK x 512         (524288)
  float* xz   = xn  + 524288;       // NTOK x 2048        (2097152)
  float* u    = xz  + 2097152;      // NTOK x 1024        (1048576)
  float* xdbl = u   + 1048576;      // NTOK x 64          (65536)
  float* dt   = xdbl+ 65536;        // NTOK x 1024        (1048576)
  float* yb   = dt  + 1048576;      // NTOK x 1024        (1048576)
  float* hn   = yb  + 1048576;      // NTOK x 512         (524288)
  float* cv   = hn  + 524288;       // 4 x 512            (2048)
  float* c1   = cv  + 2048;         // 4 x 512            (2048)

  // patch embed: im2row + WMMA GEMM (M=1024, N=512, K=256) + bias
  im2row_kernel<<<1024, 256, 0, stream>>>(x, Ap);
  launch_gemm<1, false>(Ap, 256, patch_w, 256, patch_b, h, D_MODEL,
                        NTOK, D_MODEL, 256, stream);

  for (int i = 0; i < DEPTH; ++i) {
    layernorm_kernel<<<128, 256, 0, stream>>>(h, norm_w + i * 512, norm_b + i * 512, xn, NTOK);
    // in_proj: (1024,512)x(2048,512)^T -> xz (1024,2048)
    launch_gemm<0, false>(xn, D_MODEL, in_proj_w + (size_t)i * 2048 * 512, D_MODEL,
                          nullptr, xz, 2 * D_INNER, NTOK, 2 * D_INNER, D_MODEL, stream);
    // ---- forward direction ----
    conv_silu_kernel<<<4096, 256, 0, stream>>>(xz, conv_w_f + i * 4096, conv_b_f + i * 1024, u, 0);
    launch_gemm<0, false>(u, D_INNER, xproj_w_f + i * 65536, D_INNER,
                          nullptr, xdbl, 64, NTOK, 64, D_INNER, stream);
    launch_gemm<2, false>(xdbl, 64, dtproj_w_f + i * 32768, DT_RANK,
                          dtproj_b_f + i * 1024, dt, D_INNER, NTOK, D_INNER, DT_RANK, stream);
    scan_kernel<false><<<256, 256, 0, stream>>>(dt, u, xdbl, A_log_f + i * 16384, D_f + i * 1024, yb, 0);
    // ---- backward direction (reversed window + reversed recurrence) ----
    conv_silu_kernel<<<4096, 256, 0, stream>>>(xz, conv_w_b + i * 4096, conv_b_b + i * 1024, u, 1);
    launch_gemm<0, false>(u, D_INNER, xproj_w_b + i * 65536, D_INNER,
                          nullptr, xdbl, 64, NTOK, 64, D_INNER, stream);
    launch_gemm<2, false>(xdbl, 64, dtproj_w_b + i * 32768, DT_RANK,
                          dtproj_b_b + i * 1024, dt, D_INNER, NTOK, D_INNER, DT_RANK, stream);
    scan_kernel<true><<<256, 256, 0, stream>>>(dt, u, xdbl, A_log_b + i * 16384, D_b + i * 1024, yb, 1);
    // gate + residual out_proj
    gate_kernel<<<4096, 256, 0, stream>>>(yb, xz);
    launch_gemm<0, true>(yb, D_INNER, out_proj_w + (size_t)i * 512 * 1024, D_INNER,
                         nullptr, h, D_MODEL, NTOK, D_MODEL, D_INNER, stream);
  }

  // final norm, mean-pool + classifier norm, MLP head
  layernorm_kernel<<<128, 256, 0, stream>>>(h, normf_w, normf_b, hn, NTOK);
  pool_ln_kernel<<<BATCH, 256, 0, stream>>>(hn, ln_w, ln_b, cv);
  launch_gemm<3, false>(cv, D_MODEL, fc1_w, D_MODEL, fc1_b, c1, 512, BATCH, 512, D_MODEL, stream);
  launch_gemm<1, false>(c1, 512, fc2_w, 512, fc2_b, (float*)d_out, 10, BATCH, 10, 512, stream);
}
